// MNISTAutoencoderSlimTik_37099927503137
// MI455X (gfx1250) — matrix operand order repaired and checked by reading, not compiled, not run on hardware
//
#include <hip/hip_runtime.h>
#include <cstdint>

// ---------------------------------------------------------------------------
// MNIST autoencoder forward (N=512) for MI455X / gfx1250, wave32.
//
// Roofline: output Z is 414 MB (write-once, > L2 192MB) -> ~18us floor at
// 23.3 TB/s. All compute stages total ~1 GFLOP / <100MB. Strategy:
//   * Z streamed with non-temporal stores (don't pollute L2).
//   * GEMM-shaped layers (enc2 im2col-GEMM, enc3, dec0) via
//     v_wmma_f32_16x16x32_bf16 (bf16 in, f32 accumulate).
//   * Parity-dependent conv / convT layers (enc1, dec1) as fused VALU kernels
//     with bias/ReLU/BatchNorm folded in.
// ---------------------------------------------------------------------------

typedef __bf16 bf16;
typedef __attribute__((ext_vector_type(16))) bf16  v16bf;
typedef __attribute__((ext_vector_type(8)))  bf16  v8bf;
typedef __attribute__((ext_vector_type(8)))  float v8f;

#define NB    512
#define EPS   1e-5f

__device__ __forceinline__ bf16 f2bf(float x) { return (bf16)x; }

// ---------------------------------------------------------------------------
// enc1: conv 1->16, k4 s2 p1, 28x28 -> 14x14, +bias, ReLU.  (N,16,14,14) f32
// ---------------------------------------------------------------------------
__global__ void k_enc1(const float* __restrict__ x, const float* __restrict__ w,
                       const float* __restrict__ b, float* __restrict__ h1) {
    int idx = blockIdx.x * blockDim.x + threadIdx.x;           // N*16*196
    if (idx >= NB * 16 * 196) return;
    int ow = idx % 14, oh = (idx / 14) % 14, co = (idx / 196) % 16, n = idx / 3136;
    float acc = b[co];
    #pragma unroll
    for (int i = 0; i < 4; ++i) {
        int h = 2 * oh - 1 + i;
        if (h < 0 || h >= 28) continue;
        #pragma unroll
        for (int j = 0; j < 4; ++j) {
            int ww = 2 * ow - 1 + j;
            if (ww < 0 || ww >= 28) continue;
            acc += x[n * 784 + h * 28 + ww] * w[co * 16 + i * 4 + j];
        }
    }
    h1[idx] = fmaxf(acc, 0.0f);
}

// ---------------------------------------------------------------------------
// im2col for enc2: A2[(n,oh,ow)][ci*16+i*4+j] (25088 x 256) bf16
// ---------------------------------------------------------------------------
__global__ void k_im2col2(const float* __restrict__ h1, bf16* __restrict__ A2) {
    int idx = blockIdx.x * blockDim.x + threadIdx.x;           // 25088*256
    if (idx >= 25088 * 256) return;
    int k = idx & 255, row = idx >> 8;
    int n = row / 49, s = row % 49, oh = s / 7, ow = s % 7;
    int ci = k >> 4, i = (k >> 2) & 3, j = k & 3;
    int h = 2 * oh - 1 + i, w = 2 * ow - 1 + j;
    float v = 0.0f;
    if (h >= 0 && h < 14 && w >= 0 && w < 14)
        v = h1[((n * 16 + ci) * 14 + h) * 14 + w];
    A2[idx] = f2bf(v);
}

// ---------------------------------------------------------------------------
// f32 -> bf16 repack with zero padding: dst (Rd x Cd), src (Rs x Cs) row-major
// ---------------------------------------------------------------------------
__global__ void k_pack(const float* __restrict__ src, bf16* __restrict__ dst,
                       int Rd, int Rs, int Cd, int Cs) {
    int idx = blockIdx.x * blockDim.x + threadIdx.x;
    if (idx >= Rd * Cd) return;
    int r = idx / Cd, c = idx % Cd;
    float v = (r < Rs && c < Cs) ? src[r * Cs + c] : 0.0f;
    dst[idx] = f2bf(v);
}

// ---------------------------------------------------------------------------
// WMMA GEMM: C[MxNc] = A[MxK] * BT[NcxK]^T.  bf16 in, f32 out.
// 128 threads = 4 waves, one 16x16 tile per wave. K % 32 == 0, M,Nc % 16 == 0.
// A/B k order here is contiguous, k = 16*(lane/16) + s; C layout per ISA (wave32).
// ---------------------------------------------------------------------------
__global__ void k_gemm(const bf16* __restrict__ A, const bf16* __restrict__ BT,
                       float* __restrict__ C, int M, int Nc, int K) {
    int wave = threadIdx.x >> 5;
    int lane = threadIdx.x & 31;
    int numNT = Nc >> 4;
    int tile  = blockIdx.x * 4 + wave;
    if (tile >= (M >> 4) * numNT) return;                      // wave-uniform
    int nT = tile % numNT, mT = tile / numNT;

    int arow   = mT * 16 + (lane & 15);
    int akoff  = (lane >> 4) * 16;                             // A half-select
    int bcol   = nT * 16 + (lane & 15);
    int bkoff  = (lane >> 4) * 16;                             // B half-select
    const bf16* arp = A  + (size_t)arow * K;
    const bf16* brp = BT + (size_t)bcol * K;

    v8f acc = {0.f, 0.f, 0.f, 0.f, 0.f, 0.f, 0.f, 0.f};
    for (int k0 = 0; k0 < K; k0 += 32) {
        union { v16bf v; v8bf h[2]; } a;
        a.h[0] = *(const v8bf*)(arp + k0 + akoff);             // K in [0,8)+half
        a.h[1] = *(const v8bf*)(arp + k0 + 8 + akoff);         // K in [8,16)+half
        v16bf b = *(const v16bf*)(brp + k0 + bkoff);           // 16 contiguous K
        acc = __builtin_amdgcn_wmma_f32_16x16x32_bf16(
            false, a.v, false, b, (short)0, acc, false, false);
    }

    int row0 = mT * 16 + (lane >> 4) * 8;
    union { v8f v; float f[8]; } uc; uc.v = acc;
    #pragma unroll
    for (int v = 0; v < 8; ++v)
        C[(size_t)(row0 + v) * Nc + bcol] = uc.f[v];
}

// ---------------------------------------------------------------------------
// enc2 epilogue: C2[(n,s)][co] + b, ReLU -> A3[n][co*49+s] bf16 (512x1568)
// ---------------------------------------------------------------------------
__global__ void k_h2(const float* __restrict__ C2, const float* __restrict__ b,
                     bf16* __restrict__ A3) {
    int idx = blockIdx.x * blockDim.x + threadIdx.x;           // 25088*32
    if (idx >= 25088 * 32) return;
    int col = idx & 31, row = idx >> 5;
    int n = row / 49, s = row % 49;
    A3[n * 1568 + col * 49 + s] = f2bf(fmaxf(C2[idx] + b[col], 0.0f));
}

// ---------------------------------------------------------------------------
// enc3 epilogue: C3 (512x64) + bias (cols<50) -> A4 bf16 (512x64, padded)
// ---------------------------------------------------------------------------
__global__ void k_h3(const float* __restrict__ C3, const float* __restrict__ b,
                     bf16* __restrict__ A4) {
    int idx = blockIdx.x * blockDim.x + threadIdx.x;           // 512*64
    if (idx >= 512 * 64) return;
    int c = idx & 63;
    float v = (c < 50) ? (C3[idx] + b[c]) : 0.0f;
    A4[idx] = f2bf(v);
}

// ---------------------------------------------------------------------------
// dec0 epilogue: C4 (512x1568) + bias, BatchNorm1 -> bn1h f32 (== N,32,7,7)
// ---------------------------------------------------------------------------
__global__ void k_bn1(const float* __restrict__ C4, const float* __restrict__ b,
                      const float* __restrict__ g, const float* __restrict__ bb,
                      const float* __restrict__ m, const float* __restrict__ v,
                      float* __restrict__ o) {
    int idx = blockIdx.x * blockDim.x + threadIdx.x;           // 512*1568
    if (idx >= 512 * 1568) return;
    int f = idx % 1568, c = f / 49;
    float scale = g[c] * rsqrtf(v[c] + EPS);
    o[idx] = (C4[idx] + b[f] - m[c]) * scale + bb[c];
}

// ---------------------------------------------------------------------------
// dec1: convT 32->16, k4 s2 p1, 7x7 -> 14x14, +bias, ReLU, BatchNorm2 -> feat
// ---------------------------------------------------------------------------
__global__ void k_dec1(const float* __restrict__ xin, const float* __restrict__ w,
                       const float* __restrict__ b,
                       const float* __restrict__ g, const float* __restrict__ bb,
                       const float* __restrict__ m, const float* __restrict__ v,
                       float* __restrict__ feat) {
    int idx = blockIdx.x * blockDim.x + threadIdx.x;           // 512*16*196
    if (idx >= NB * 16 * 196) return;
    int W = idx % 14, H = (idx / 14) % 14, co = (idx / 196) % 16, n = idx / 3136;
    float acc = b[co];
    #pragma unroll
    for (int i = 0; i < 4; ++i) {
        int uh = H + 1 - i;
        if (uh < 0 || (uh & 1) || uh >= 14) continue;          // h = uh/2 in [0,7)
        int h = uh >> 1;
        #pragma unroll
        for (int j = 0; j < 4; ++j) {
            int uw = W + 1 - j;
            if (uw < 0 || (uw & 1) || uw >= 14) continue;
            int ww = uw >> 1;
            const float* ap = xin + n * 1568 + h * 7 + ww;     // + ci*49
            const float* wp = w + co * 16 + i * 4 + j;         // + ci*256
            #pragma unroll
            for (int ci = 0; ci < 32; ++ci)
                acc += ap[ci * 49] * wp[ci * 256];
        }
    }
    acc = fmaxf(acc, 0.0f);
    float scale = g[co] * rsqrtf(v[co] + EPS);
    feat[idx] = (acc - m[co]) * scale + bb[co];
}

// ---------------------------------------------------------------------------
// final out: convT 16->1 with Wf + bf -> (N,1,28,28)
// ---------------------------------------------------------------------------
__global__ void k_out(const float* __restrict__ feat, const float* __restrict__ w,
                      const float* __restrict__ bf_, float* __restrict__ out) {
    int idx = blockIdx.x * blockDim.x + threadIdx.x;           // 512*784
    if (idx >= NB * 784) return;
    int W = idx % 28, H = (idx / 28) % 28, n = idx / 784;
    float acc = bf_[0];
    #pragma unroll
    for (int i = 0; i < 4; ++i) {
        int uh = H + 1 - i;
        if (uh < 0 || (uh & 1) || uh >= 28) continue;
        int h = uh >> 1;
        #pragma unroll
        for (int j = 0; j < 4; ++j) {
            int uw = W + 1 - j;
            if (uw < 0 || (uw & 1) || uw >= 28) continue;
            int ww = uw >> 1;
            #pragma unroll
            for (int c = 0; c < 16; ++c)
                acc += feat[((n * 16 + c) * 14 + h) * 14 + ww] * w[c * 16 + i * 4 + j];
        }
    }
    __builtin_nontemporal_store(acc, out + idx);
}

// ---------------------------------------------------------------------------
// Z builder: one 256-thread block per row (n,H,W); col = c*16+i*4+j, col 256=1.
// 414 MB write-once stream -> non-temporal stores (bypass 192MB L2 rinse).
// feat[n] (12.5KB) is reused across the 784 rows of image n -> stays cached.
// ---------------------------------------------------------------------------
__global__ void k_z(const float* __restrict__ feat, float* __restrict__ Z) {
    int r = blockIdx.x;                                        // N*784 rows
    int n = r / 784, pix = r % 784, H = pix / 28, W = pix % 28;
    int t = threadIdx.x;                                       // 256 threads
    int c = t >> 4, i = (t >> 2) & 3, j = t & 3;
    int uh = H + 1 - i, uw = W + 1 - j;
    float val = 0.0f;
    if (uh >= 0 && uw >= 0 && !(uh & 1) && !(uw & 1) && uh < 28 && uw < 28)
        val = feat[((n * 16 + c) * 14 + (uh >> 1)) * 14 + (uw >> 1)];
    float* zr = Z + (size_t)r * 257;
    __builtin_nontemporal_store(val, zr + t);
    if (t == 0) __builtin_nontemporal_store(1.0f, zr + 256);
}

// ---------------------------------------------------------------------------
extern "C" void kernel_launch(void* const* d_in, const int* in_sizes, int n_in,
                              void* d_out, int out_size, void* d_ws, size_t ws_size,
                              hipStream_t stream) {
    const float* x      = (const float*)d_in[0];
    const float* enc1_w = (const float*)d_in[1];
    const float* enc1_b = (const float*)d_in[2];
    const float* enc2_w = (const float*)d_in[3];
    const float* enc2_b = (const float*)d_in[4];
    const float* enc3_w = (const float*)d_in[5];
    const float* enc3_b = (const float*)d_in[6];
    const float* dec0_w = (const float*)d_in[7];
    const float* dec0_b = (const float*)d_in[8];
    const float* bn1_g  = (const float*)d_in[9];
    const float* bn1_b  = (const float*)d_in[10];
    const float* bn1_m  = (const float*)d_in[11];
    const float* bn1_v  = (const float*)d_in[12];
    const float* dec1_w = (const float*)d_in[13];
    const float* dec1_b = (const float*)d_in[14];
    const float* bn2_g  = (const float*)d_in[15];
    const float* bn2_b  = (const float*)d_in[16];
    const float* bn2_m  = (const float*)d_in[17];
    const float* bn2_v  = (const float*)d_in[18];
    const float* Wf     = (const float*)d_in[19];
    const float* bfc    = (const float*)d_in[20];

    float* out = (float*)d_out;                  // (512,1,28,28) = 401408
    float* Z   = out + (size_t)NB * 784;         // (512*784, 257)

    // workspace carve-out (256B aligned slabs)
    uint8_t* wsb = (uint8_t*)d_ws;
    size_t off = 0;
    auto alloc = [&](size_t bytes) -> void* {
        off = (off + 255) & ~(size_t)255;
        void* p = wsb + off; off += bytes; return p;
    };
    float* h1   = (float*)alloc((size_t)NB * 16 * 196 * 4);    // enc1 out
    bf16*  A2   = (bf16*) alloc((size_t)25088 * 256 * 2);      // enc2 im2col
    bf16*  B2T  = (bf16*) alloc((size_t)32 * 256 * 2);
    float* C2   = (float*)alloc((size_t)25088 * 32 * 4);
    bf16*  A3   = (bf16*) alloc((size_t)512 * 1568 * 2);       // enc3 A
    bf16*  B3T  = (bf16*) alloc((size_t)64 * 1568 * 2);
    float* C3   = (float*)alloc((size_t)512 * 64 * 4);
    bf16*  A4   = (bf16*) alloc((size_t)512 * 64 * 2);         // dec0 A
    bf16*  B4T  = (bf16*) alloc((size_t)1568 * 64 * 2);
    float* C4   = (float*)alloc((size_t)512 * 1568 * 4);
    float* bn1h = (float*)alloc((size_t)512 * 1568 * 4);
    float* feat = (float*)alloc((size_t)NB * 16 * 196 * 4);

    auto blocks = [](size_t n) { return (unsigned)((n + 255) / 256); };

    // Stage 1: enc1 conv + ReLU
    k_enc1<<<blocks((size_t)NB * 16 * 196), 256, 0, stream>>>(x, enc1_w, enc1_b, h1);

    // Stage 2: enc2 as implicit GEMM (25088 x 32 x 256), WMMA bf16
    k_im2col2<<<blocks((size_t)25088 * 256), 256, 0, stream>>>(h1, A2);
    k_pack<<<blocks((size_t)32 * 256), 256, 0, stream>>>(enc2_w, B2T, 32, 32, 256, 256);
    { int T = (25088 / 16) * (32 / 16);
      k_gemm<<<(T + 3) / 4, 128, 0, stream>>>(A2, B2T, C2, 25088, 32, 256); }
    k_h2<<<blocks((size_t)25088 * 32), 256, 0, stream>>>(C2, enc2_b, A3);

    // Stage 3: enc3 linear (512 x 64pad x 1568), WMMA bf16
    k_pack<<<blocks((size_t)64 * 1568), 256, 0, stream>>>(enc3_w, B3T, 64, 50, 1568, 1568);
    { int T = (512 / 16) * (64 / 16);
      k_gemm<<<(T + 3) / 4, 128, 0, stream>>>(A3, B3T, C3, 512, 64, 1568); }
    k_h3<<<blocks((size_t)512 * 64), 256, 0, stream>>>(C3, enc3_b, A4);

    // Stage 4: dec0 linear (512 x 1568 x 64pad), WMMA bf16, then bias+BN1
    k_pack<<<blocks((size_t)1568 * 64), 256, 0, stream>>>(dec0_w, B4T, 1568, 1568, 64, 50);
    { int T = (512 / 16) * (1568 / 16);
      k_gemm<<<(T + 3) / 4, 128, 0, stream>>>(A4, B4T, C4, 512, 1568, 64); }
    k_bn1<<<blocks((size_t)512 * 1568), 256, 0, stream>>>(C4, dec0_b, bn1_g, bn1_b,
                                                          bn1_m, bn1_v, bn1h);

    // Stage 5: dec1 convT + bias + ReLU + BN2 -> feat
    k_dec1<<<blocks((size_t)NB * 16 * 196), 256, 0, stream>>>(bn1h, dec1_w, dec1_b,
                                                              bn2_g, bn2_b, bn2_m, bn2_v,
                                                              feat);

    // Stage 6: outputs (final convT + bf, and the 414MB Z stream)
    k_out<<<blocks((size_t)NB * 784), 256, 0, stream>>>(feat, Wf, bfc, out);
    k_z<<<NB * 784, 256, 0, stream>>>(feat, Z);

    (void)in_sizes; (void)n_in; (void)out_size; (void)ws_size;
}
